// CrossAttention_Video_Partially_11519102288122
// MI455X (gfx1250) — compile-verified, matrix operand order checked by reading, not run on hardware
//
#include <hip/hip_runtime.h>
#include <hip/hip_bf16.h>
#include <type_traits>

// ---------------------------------------------------------------------------
// CDNA5 (gfx1250) windowed cross-attention, bf16 WMMA pipeline + TDM staging.
// ---------------------------------------------------------------------------

typedef __attribute__((ext_vector_type(16))) __bf16 v16bf;
typedef __attribute__((ext_vector_type(8)))  float  v8f;
typedef __attribute__((ext_vector_type(4)))  unsigned v4u;
typedef __attribute__((ext_vector_type(4)))  int      v4i;
typedef __attribute__((ext_vector_type(8)))  int      v8i;

#if __has_builtin(__builtin_amdgcn_tensor_load_to_lds) && \
    __has_builtin(__builtin_amdgcn_s_wait_tensorcnt)
#define ATHENA_HAVE_TDM 1
#else
#define ATHENA_HAVE_TDM 0
#endif

union Frag16 {
    v16bf    v;
    unsigned u[8];
};

__device__ __forceinline__ v8f wmma_bf16f32(const Frag16& a, const Frag16& b, v8f c) {
    return __builtin_amdgcn_wmma_f32_16x16x32_bf16(false, a.v, false, b.v,
                                                   (short)0, c, false, false);
}

// Butterfly step via ds_swizzle_b32, group-of-32 XOR mode (and=0x1F, or=0, xor=m).
#define SWZ_XOR_F32(v, m) \
    __uint_as_float((unsigned)__builtin_amdgcn_ds_swizzle((int)__float_as_uint(v), \
                                                          0x1F | ((m) << 10)))

// --------------------------- TDM 2D tile -> LDS ----------------------------
// Loads a tile of bf16 (data_size=2B): tile_w elements per row, tile_h rows,
// row stride = stride_elems, with LDS padding pad_amount DWORDs inserted
// after every pad_interval DWORDs (to match padded LDS row pitch).
// pad_interval_code: interval = 2^(code+1) DWORDs; pad_amount_code: code+1 DW.
__device__ __forceinline__ void tdm_load_2d_bf16(unsigned lds_off, const void* gptr,
                                                 unsigned tile_w, unsigned tile_h,
                                                 unsigned stride_elems,
                                                 unsigned pad_interval_code,
                                                 unsigned pad_amount_code) {
#if ATHENA_HAVE_TDM
    unsigned long long ga = (unsigned long long)gptr;
    v4u g0;
    g0[0] = 1u;                                  // count=1, user descriptor
    g0[1] = lds_off;                             // LDS byte address
    g0[2] = (unsigned)ga;                        // global_addr[31:0]
    g0[3] = (unsigned)(ga >> 32) | (2u << 30);   // global_addr[56:32], type=2
    v8i g1;
    g1[0] = (int)((1u << 16)                     // data_size = 2 bytes
                | (1u << 20)                     // pad_enable
                | (pad_interval_code << 22)
                | (pad_amount_code << 25));      // workgroup_mask = 0 (no cluster)
    g1[1] = (int)((tile_w & 0xFFFFu) << 16);                 // tensor_dim0[15:0]
    g1[2] = (int)(((tile_w >> 16) & 0xFFFFu)                 // tensor_dim0[31:16]
                | ((tile_h & 0xFFFFu) << 16));               // tensor_dim1[15:0]
    g1[3] = (int)(((tile_h >> 16) & 0xFFFFu)                 // tensor_dim1[31:16]
                | ((tile_w & 0xFFFFu) << 16));               // tile_dim0
    g1[4] = (int)(tile_h & 0xFFFFu);                         // tile_dim1, tile_dim2=0
    g1[5] = (int)stride_elems;                               // tensor_dim0_stride lo
    g1[6] = 0;                                               // stride hi, dim1_stride lo
    g1[7] = 0;
    v4i z4 = {0, 0, 0, 0};
#if defined(__clang_major__) && (__clang_major__ >= 23)
    v8i z8 = {0, 0, 0, 0, 0, 0, 0, 0};
    __builtin_amdgcn_tensor_load_to_lds(g0, g1, z4, z4, z8, 0);
#else
    __builtin_amdgcn_tensor_load_to_lds(g0, g1, z4, z4, 0);
#endif
#else
    (void)lds_off; (void)gptr; (void)tile_w; (void)tile_h;
    (void)stride_elems; (void)pad_interval_code; (void)pad_amount_code;
#endif
}

__device__ __forceinline__ void tdm_wait() {
#if ATHENA_HAVE_TDM
    __builtin_amdgcn_s_wait_tensorcnt(0);
#endif
}

// ------------------------------ prep kernels -------------------------------

__global__ __launch_bounds__(256) void cvt_f32_bf16_kernel(
        const float* __restrict__ src, __hip_bfloat16* __restrict__ dst, int n) {
    int i = blockIdx.x * 256 + threadIdx.x;
    if (i < n) dst[i] = __float2bfloat16(src[i]);
}

// hc[row][d] = hint[row][d] + pe(row % 256, d), D = 1024
__global__ __launch_bounds__(256) void hc_pe_kernel(
        const float* __restrict__ hint, float* __restrict__ hc, int rows, int D) {
    int idx = blockIdx.x * 256 + threadIdx.x;
    if (idx >= rows * D) return;
    int d   = idx % D;
    int pos = (idx / D) % 256;   // NK = 256
    int i2  = d >> 1;
    float ang = (float)pos * __powf(10000.0f, -2.0f * (float)i2 / (float)D);
    float pe  = (d & 1) ? __cosf(ang) : __sinf(ang);
    hc[idx] = hint[idx] + pe;
}

// ------------------------------ GEMM kernel --------------------------------
// C[M,N] = A[M,K] * B[K,N] (+bias). A: f32 (converted while staging) or bf16
// (staged by TDM with pad 32->40). B: bf16 row-major, transposed into LDS.
// BM=BN=128, BK=32, 8 waves; each wave owns 32x64 = 2x4 WMMA accumulators.

__device__ __forceinline__ void stage4(const float* g, __hip_bfloat16* s) {
    float4 f = *(const float4*)g;
    s[0] = __float2bfloat16(f.x);
    s[1] = __float2bfloat16(f.y);
    s[2] = __float2bfloat16(f.z);
    s[3] = __float2bfloat16(f.w);
}
__device__ __forceinline__ void stage4(const __hip_bfloat16* g, __hip_bfloat16* s) {
    *(uint2*)s = *(const uint2*)g;
}
__device__ __forceinline__ void storeC(float* p, float v) { *p = v; }
__device__ __forceinline__ void storeC(__hip_bfloat16* p, float v) { *p = __float2bfloat16(v); }

template <typename AT, typename CT>
__global__ __launch_bounds__(256) void gemm_bf16_wmma_kernel(
        const AT* __restrict__ A, const __hip_bfloat16* __restrict__ B,
        CT* __restrict__ C, const float* __restrict__ bias,
        int M, int N, int K) {
    constexpr int BM = 128, BN = 128, BK = 32, KP = 40;  // KP: 80B row = 20 banks
    constexpr bool kUseTdmA =
        (ATHENA_HAVE_TDM != 0) && std::is_same<AT, __hip_bfloat16>::value;
    __shared__ __hip_bfloat16 As[BM][KP];
    __shared__ __hip_bfloat16 Bs[BN][KP];   // B stored transposed: [n][k]

    const int tid  = threadIdx.x;
    const int lane = tid & 31;
    const int wave = tid >> 5;       // 0..7
    const int wr   = wave >> 1;      // 0..3  (wave row: 32 rows each)
    const int wc   = wave & 1;       // 0..1  (wave col: 64 cols each)
    const int half = lane >> 4;
    const int nl   = lane & 15;
    const long m0  = (long)blockIdx.x * BM;
    const long n0  = (long)blockIdx.y * BN;

    const v8f vzero = {0.f, 0.f, 0.f, 0.f, 0.f, 0.f, 0.f, 0.f};
    v8f acc[2][4];
#pragma unroll
    for (int i = 0; i < 2; i++)
#pragma unroll
        for (int j = 0; j < 4; j++) acc[i][j] = vzero;

    for (int kb = 0; kb < K; kb += BK) {
        // ---- stage A tile [128 x 32] ----
        if constexpr (kUseTdmA) {
            // TDM: direct bf16 copy with pad 64B-row -> 80B LDS pitch.
            // interval 16 DW (code 3), pad 4 DW (code 3).
            if (wave == 0)
                tdm_load_2d_bf16((unsigned)(size_t)&As[0][0],
                                 A + m0 * (long)K + kb, BK, BM, (unsigned)K, 3, 3);
        } else {
#pragma unroll
            for (int i = 0; i < 4; i++) {
                int idx = tid + i * 256;
                int row = idx >> 3;
                int kq  = (idx & 7) << 2;
                const AT* gp = A + (m0 + row) * (long)K + kb + kq;
                stage4(gp, &As[row][kq]);
                if (kb + BK < K) __builtin_prefetch(gp + BK, 0, 1);
            }
        }
        // ---- stage B tile [32 x 128], transposed into LDS [n][k] ----
#pragma unroll
        for (int i = 0; i < 4; i++) {
            int idx = tid + i * 256;
            int kk  = idx >> 5;               // 0..31
            int n4  = (idx & 31) << 2;        // 0..124
            __hip_bfloat16 t[4];
            const __hip_bfloat16* gp = B + (long)(kb + kk) * N + n0 + n4;
            *(uint2*)t = *(const uint2*)gp;
            Bs[n4 + 0][kk] = t[0];
            Bs[n4 + 1][kk] = t[1];
            Bs[n4 + 2][kk] = t[2];
            Bs[n4 + 3][kk] = t[3];
            if (kb + BK < K) __builtin_prefetch(gp + (long)BK * N, 0, 1);
        }
        if constexpr (kUseTdmA) {
            if (wave == 0) tdm_wait();
        }
        __syncthreads();

        // ---- load fragments + 8 WMMAs ----
        Frag16 af[2], bf[4];
#pragma unroll
        for (int ms = 0; ms < 2; ms++) {
            int row = wr * 32 + ms * 16 + nl;      // lane holds row M = lane%16
#pragma unroll
            for (int r = 0; r < 8; r++) {
                int kk = ((r >> 2) << 4) + ((r & 3) << 1) + (half << 3);
                af[ms].u[r] = *(const unsigned*)&As[row][kk];
            }
        }
#pragma unroll
        for (int ns = 0; ns < 4; ns++) {
            int col = wc * 64 + ns * 16 + nl;      // lane holds col N = lane%16
#pragma unroll
            for (int r = 0; r < 8; r++) {
                int kk = ((r >> 2) << 4) + ((r & 3) << 1) + (half << 3);
                bf[ns].u[r] = *(const unsigned*)&Bs[col][kk];
            }
        }
#pragma unroll
        for (int ms = 0; ms < 2; ms++)
#pragma unroll
            for (int ns = 0; ns < 4; ns++)
                acc[ms][ns] = wmma_bf16f32(af[ms], bf[ns], acc[ms][ns]);
        __syncthreads();
    }

    // ---- epilogue: C layout -> lane nl = column, row = j + 8*half ----
#pragma unroll
    for (int ms = 0; ms < 2; ms++)
#pragma unroll
        for (int ns = 0; ns < 4; ns++) {
            long col = n0 + wc * 64 + ns * 16 + nl;
            float bb = bias ? bias[col] : 0.f;
#pragma unroll
            for (int j = 0; j < 8; j++) {
                long row = m0 + wr * 32 + ms * 16 + j + half * 8;
                storeC(&C[row * (long)N + col], acc[ms][ns][j] + bb);
            }
        }
}

// ---------------------------- attention kernel -----------------------------
// One workgroup = 128 query rows of one (b,h,w) block. 8 waves x 16 rows.
// S = Q K^T * 0.125 -> exact softmax over 64 keys -> O = P V. All bf16 WMMA.
// K tile staged by TDM (pad 64->72 elements); V^T staged manually.

__global__ __launch_bounds__(256) void attn_window_kernel(
        const __hip_bfloat16* __restrict__ q, const __hip_bfloat16* __restrict__ k,
        const __hip_bfloat16* __restrict__ v, __hip_bfloat16* __restrict__ o) {
    constexpr int CC = 512;   // inner = H*DH
    constexpr int DP = 72;    // padded 64 (144B = 36 banks, conflict-free)
    __shared__ __hip_bfloat16 Ks[64][DP];      // K tile [key][dh]
    __shared__ __hip_bfloat16 Vt[64][DP];      // V^T   [dh][key]
    __shared__ __hip_bfloat16 Ps[8][16][DP];   // per-wave P tile [row][key]

    const int tid  = threadIdx.x;
    const int lane = tid & 31;
    const int wave = tid >> 5;
    const int half = lane >> 4;
    const int nl   = lane & 15;

    int bid   = blockIdx.x;          // 0..2047
    int chunk = bid & 7;             // which 128-row group of the 1024-row window
    int bhw   = bid >> 3;
    int w     = bhw & 3;
    int h     = (bhw >> 2) & 7;
    int b     = bhw >> 5;

    const long qrow0 = (long)b * 4096 + w * 1024 + chunk * 128;
    const long krow0 = (long)b * 256 + w * 64;
    const int  col0  = h * 64;

    // ---- stage K (TDM, row-major with pad) and V^T (manual transpose) ----
#if ATHENA_HAVE_TDM
    if (wave == 0)
        // 64 elems/row (128B = 32 DW, interval code 4), pad 4 DW (code 3) -> 72
        tdm_load_2d_bf16((unsigned)(size_t)&Ks[0][0],
                         k + krow0 * CC + col0, 64, 64, CC, 4, 3);
#endif
#pragma unroll
    for (int i = 0; i < 4; i++) {
        int idx = tid + i * 256;     // 0..1023
        int key = idx >> 4;          // 0..63
        int d4  = (idx & 15) << 2;   // 0..60
#if !ATHENA_HAVE_TDM
        *(uint2*)&Ks[key][d4] = *(const uint2*)(k + (krow0 + key) * CC + col0 + d4);
#endif
        __hip_bfloat16 t[4];
        *(uint2*)t = *(const uint2*)(v + (krow0 + key) * CC + col0 + d4);
        Vt[d4 + 0][key] = t[0];
        Vt[d4 + 1][key] = t[1];
        Vt[d4 + 2][key] = t[2];
        Vt[d4 + 3][key] = t[3];
    }
#if ATHENA_HAVE_TDM
    if (wave == 0) tdm_wait();
#endif
    __syncthreads();

    // ---- Q fragments (A-matrix, 2 k-steps over DH=64), straight from HBM ----
    Frag16 aq[2];
    {
        long qrow = qrow0 + wave * 16 + nl;
#pragma unroll
        for (int ks = 0; ks < 2; ks++)
#pragma unroll
            for (int r = 0; r < 8; r++) {
                int kk = ((r >> 2) << 4) + ((r & 3) << 1) + (half << 3) + ks * 32;
                aq[ks].u[r] = *(const unsigned*)(q + qrow * CC + col0 + kk);
            }
    }

    // ---- S = Q K^T : 4 key tiles x 2 k-steps ----
    v8f s[4];
#pragma unroll
    for (int nt = 0; nt < 4; nt++) {
        Frag16 bk[2];
#pragma unroll
        for (int ks = 0; ks < 2; ks++)
#pragma unroll
            for (int r = 0; r < 8; r++) {
                int kk = ((r >> 2) << 4) + ((r & 3) << 1) + (half << 3) + ks * 32;
                bk[ks].u[r] = *(const unsigned*)&Ks[nt * 16 + nl][kk];
            }
        v8f a = {0.f, 0.f, 0.f, 0.f, 0.f, 0.f, 0.f, 0.f};
        a = wmma_bf16f32(aq[0], bk[0], a);
        a = wmma_bf16f32(aq[1], bk[1], a);
        s[nt] = a;
    }

    // ---- softmax over 64 keys; row j+8*half lives in a 16-lane half ----
    const float scale = 0.125f;   // DH^-0.5
#pragma unroll
    for (int j = 0; j < 8; j++) {
        float t0 = s[0][j] * scale, t1 = s[1][j] * scale;
        float t2 = s[2][j] * scale, t3 = s[3][j] * scale;
        float m = fmaxf(fmaxf(t0, t1), fmaxf(t2, t3));
        m = fmaxf(m, SWZ_XOR_F32(m, 8));
        m = fmaxf(m, SWZ_XOR_F32(m, 4));
        m = fmaxf(m, SWZ_XOR_F32(m, 2));
        m = fmaxf(m, SWZ_XOR_F32(m, 1));
        float e0 = __expf(t0 - m), e1 = __expf(t1 - m);
        float e2 = __expf(t2 - m), e3 = __expf(t3 - m);
        float sum = e0 + e1 + e2 + e3;
        sum += SWZ_XOR_F32(sum, 8);
        sum += SWZ_XOR_F32(sum, 4);
        sum += SWZ_XOR_F32(sum, 2);
        sum += SWZ_XOR_F32(sum, 1);
        float r = 1.0f / sum;
        int row = j + half * 8;
        Ps[wave][row][0 * 16 + nl] = __float2bfloat16(e0 * r);
        Ps[wave][row][1 * 16 + nl] = __float2bfloat16(e1 * r);
        Ps[wave][row][2 * 16 + nl] = __float2bfloat16(e2 * r);
        Ps[wave][row][3 * 16 + nl] = __float2bfloat16(e3 * r);
    }
    // per-wave LDS region, DS ops are in-order within a wave -> no barrier

    // ---- P as A-fragments ----
    Frag16 ap[2];
#pragma unroll
    for (int ks = 0; ks < 2; ks++)
#pragma unroll
        for (int r = 0; r < 8; r++) {
            int kk = ((r >> 2) << 4) + ((r & 3) << 1) + (half << 3) + ks * 32;
            ap[ks].u[r] = *(const unsigned*)&Ps[wave][nl][kk];
        }

    // ---- O = P V : 4 dh tiles x 2 k-steps over keys ----
#pragma unroll
    for (int nt = 0; nt < 4; nt++) {
        Frag16 bv[2];
#pragma unroll
        for (int ks = 0; ks < 2; ks++)
#pragma unroll
            for (int r = 0; r < 8; r++) {
                int kk = ((r >> 2) << 4) + ((r & 3) << 1) + (half << 3) + ks * 32;
                bv[ks].u[r] = *(const unsigned*)&Vt[nt * 16 + nl][kk];
            }
        v8f a = {0.f, 0.f, 0.f, 0.f, 0.f, 0.f, 0.f, 0.f};
        a = wmma_bf16f32(ap[0], bv[0], a);
        a = wmma_bf16f32(ap[1], bv[1], a);
#pragma unroll
        for (int j = 0; j < 8; j++) {
            long row = qrow0 + wave * 16 + j + half * 8;
            o[row * CC + col0 + nt * 16 + nl] = __float2bfloat16(a[j]);
        }
    }
}

// ------------------------------ host launcher ------------------------------

extern "C" void kernel_launch(void* const* d_in, const int* in_sizes, int n_in,
                              void* d_out, int out_size, void* d_ws, size_t ws_size,
                              hipStream_t stream) {
    (void)in_sizes; (void)n_in; (void)out_size; (void)ws_size;
    const float* x    = (const float*)d_in[0];  // [8,4096,1024]
    const float* hint = (const float*)d_in[1];  // [8,256,1024]
    const float* Wq   = (const float*)d_in[2];  // [1024,512]
    const float* Wk   = (const float*)d_in[3];
    const float* Wv   = (const float*)d_in[4];
    const float* Wo   = (const float*)d_in[5];  // [512,1024]
    const float* bo   = (const float*)d_in[6];  // [1024]
    float* out = (float*)d_out;                 // [8,4096,1024]

    const int Mq = 8 * 4096;   // 32768
    const int Mk = 8 * 256;    // 2048
    const int QD = 1024, INNER = 512;
    const int WN = QD * INNER; // 524288 elements per weight

    char* ws = (char*)d_ws;
    size_t off = 0;
    __hip_bfloat16* wq_bf = (__hip_bfloat16*)(ws + off); off += (size_t)WN * 2;
    __hip_bfloat16* wk_bf = (__hip_bfloat16*)(ws + off); off += (size_t)WN * 2;
    __hip_bfloat16* wv_bf = (__hip_bfloat16*)(ws + off); off += (size_t)WN * 2;
    __hip_bfloat16* wo_bf = (__hip_bfloat16*)(ws + off); off += (size_t)WN * 2;
    float*          hc    = (float*)         (ws + off); off += (size_t)Mk * QD * 4;
    __hip_bfloat16* q_bf  = (__hip_bfloat16*)(ws + off); off += (size_t)Mq * INNER * 2;
    __hip_bfloat16* k_bf  = (__hip_bfloat16*)(ws + off); off += (size_t)Mk * INNER * 2;
    __hip_bfloat16* v_bf  = (__hip_bfloat16*)(ws + off); off += (size_t)Mk * INNER * 2;
    __hip_bfloat16* a_bf  = (__hip_bfloat16*)(ws + off); off += (size_t)Mq * INNER * 2;

    // weights -> bf16
    cvt_f32_bf16_kernel<<<WN / 256, 256, 0, stream>>>(Wq, wq_bf, WN);
    cvt_f32_bf16_kernel<<<WN / 256, 256, 0, stream>>>(Wk, wk_bf, WN);
    cvt_f32_bf16_kernel<<<WN / 256, 256, 0, stream>>>(Wv, wv_bf, WN);
    cvt_f32_bf16_kernel<<<WN / 256, 256, 0, stream>>>(Wo, wo_bf, WN);
    // hc = hint + PE
    hc_pe_kernel<<<(Mk * QD) / 256, 256, 0, stream>>>(hint, hc, Mk, QD);

    // projections (bf16 WMMA, f32 accumulate)
    gemm_bf16_wmma_kernel<float, __hip_bfloat16>
        <<<dim3(Mq / 128, INNER / 128), 256, 0, stream>>>(x, wq_bf, q_bf, nullptr, Mq, INNER, QD);
    gemm_bf16_wmma_kernel<float, __hip_bfloat16>
        <<<dim3(Mk / 128, INNER / 128), 256, 0, stream>>>(hc, wk_bf, k_bf, nullptr, Mk, INNER, QD);
    gemm_bf16_wmma_kernel<float, __hip_bfloat16>
        <<<dim3(Mk / 128, INNER / 128), 256, 0, stream>>>(hc, wv_bf, v_bf, nullptr, Mk, INNER, QD);

    // windowed attention: 8*8*4 blocks x 8 row-chunks = 2048 workgroups
    attn_window_kernel<<<2048, 256, 0, stream>>>(q_bf, k_bf, v_bf, a_bf);

    // output projection + bias -> f32 (A staged via TDM)
    gemm_bf16_wmma_kernel<__hip_bfloat16, float>
        <<<dim3(Mq / 128, QD / 128), 256, 0, stream>>>(a_bf, wo_bf, out, bo, Mq, QD, INNER);
}
